// MBertEncoder_5239860101846
// MI455X (gfx1250) — matrix operand order repaired and checked by reading, not compiled, not run on hardware
//
#include <hip/hip_runtime.h>
#include <hip/hip_bf16.h>

#define LYR 12
#define H   768
#define NH  12
#define HD  64
#define FFD 3072
#define BB  8
#define TT  512
#define BT  (BB*TT)          // 4096
#define ATT_SCALE 0.125f     // 1/sqrt(64)

typedef __bf16 bf_t;
typedef __attribute__((ext_vector_type(16))) __bf16 v16bf;
typedef __attribute__((ext_vector_type(8)))  float  v8f;
typedef __attribute__((ext_vector_type(4)))  int    v4i;

union Frag16 { v16bf v; uint4 q[2]; };
union Pack8  { uint4 q; bf_t h[8]; };

__device__ __forceinline__ v8f wmma_bf16(const Frag16& a, const Frag16& b, v8f c) {
  // D = A(16x32 bf16) * B(32x16 bf16) + C(16x16 f32)
  return __builtin_amdgcn_wmma_f32_16x16x32_bf16(false, a.v, false, b.v,
                                                 (short)0, c, false, false);
}

// CDNA5 async global->LDS copy (no VGPR transit, tracked on ASYNCcnt).
// Builtin signature (from compiler diagnostic): (v4i addrspace(1)*, v4i addrspace(3)*, int, int)
#if defined(__AMDGCN__) && __has_builtin(__builtin_amdgcn_global_load_async_to_lds_b128)
  #define ASYNC_A 1
  typedef __attribute__((address_space(1))) v4i ga_v4i;   // global int4*
  typedef __attribute__((address_space(3))) v4i ls_v4i;   // LDS int4*
  __device__ __forceinline__ void async_wait0() {
  #if __has_builtin(__builtin_amdgcn_s_wait_asynccnt)
    __builtin_amdgcn_s_wait_asynccnt(0);
  #else
    asm volatile("s_wait_asynccnt 0x0" ::: "memory");
  #endif
  }
#else
  #define ASYNC_A 0
#endif

// ---------------------------------------------------------------- fp32 -> bf16
__global__ __launch_bounds__(256) void cvt_kernel(const float* __restrict__ s,
                                                  bf_t* __restrict__ d, size_t n) {
  size_t i = (size_t)blockIdx.x * 256 + threadIdx.x;
  if (i < n) d[i] = (bf_t)s[i];
}

// ---------------------------------------------------------------- bf16 GEMM
// Block tile 256(M) x 128(N), k-step 32. 8 waves; each wave owns 64x64:
// 4x4 = 16 v_wmma_f32_16x16x32_bf16 per k-step from 8 fragment-pair LDS loads.
// C[M,N] = A[M,K] * W[K,N] + bias, epilogues:
//   epi==0: out bf16        epi==1: +residual, out f32       epi==2: GELU, out bf16
__global__ __launch_bounds__(256) void gemm_bf16_kernel(
    const bf_t* __restrict__ A, const bf_t* __restrict__ W,
    const float* __restrict__ bias, const float* __restrict__ residual,
    float* __restrict__ outF, bf_t* __restrict__ outB,
    int M, int N, int K, int epi)
{
  __shared__ __align__(16) bf_t Alds[256 * 40];   // 256 rows x 32 k (pad 40)
  __shared__ __align__(16) bf_t Blds[128 * 40];   // transposed: 128 n x 32 k (pad 40)

  const int tid  = threadIdx.x;
  const int wid  = tid >> 5;
  const int lane = tid & 31;
  const int l16  = lane & 15;
  const int lh   = lane >> 4;        // which half of the wave
  const int waveM = wid >> 1;        // 0..3 -> 64 rows each
  const int waveN = wid & 1;         // 0..1 -> 64 cols each
  const int bm = blockIdx.y * 256;
  const int bn = blockIdx.x * 128;

  v8f acc[4][4];
  #pragma unroll
  for (int i = 0; i < 4; ++i)
    #pragma unroll
    for (int j = 0; j < 4; ++j)
      #pragma unroll
      for (int r = 0; r < 8; ++r) acc[i][j][r] = 0.0f;

  const int arow = bm + tid;         // one full 32-half row per thread
  const int bk   = tid >> 3;         // 0..31 (k within tile)
  const int bseg = tid & 7;          // 16-elem segment along n

  const int KT = K >> 5;
  for (int kt = 0; kt < KT; ++kt) {
    { // A tile 256x32: one row (32 halves = 64B) per thread
      const bf_t* src = A + (size_t)arow * K + kt * 32;
      bf_t* dst = &Alds[tid * 40];
#if ASYNC_A
      // IOFFSET applies to both global and LDS address -> 4 async b128 chunks
      __builtin_amdgcn_global_load_async_to_lds_b128((ga_v4i*)src, (ls_v4i*)dst, 0,  0);
      __builtin_amdgcn_global_load_async_to_lds_b128((ga_v4i*)src, (ls_v4i*)dst, 16, 0);
      __builtin_amdgcn_global_load_async_to_lds_b128((ga_v4i*)src, (ls_v4i*)dst, 32, 0);
      __builtin_amdgcn_global_load_async_to_lds_b128((ga_v4i*)src, (ls_v4i*)dst, 48, 0);
#else
      uint4 a0 = *(const uint4*)(src);
      uint4 a1 = *(const uint4*)(src + 8);
      uint4 a2 = *(const uint4*)(src + 16);
      uint4 a3 = *(const uint4*)(src + 24);
      *(uint4*)(dst)      = a0;
      *(uint4*)(dst + 8)  = a1;
      *(uint4*)(dst + 16) = a2;
      *(uint4*)(dst + 24) = a3;
#endif
      if (kt + 1 < KT) __builtin_prefetch(src + 32, 0, 1);
    }
    { // W tile 32x128: 16 halves per thread, transposed into Blds[n][k]
      const bf_t* src = W + (size_t)(kt * 32 + bk) * N + bn + bseg * 16;
      Pack8 p0, p1;
      p0.q = *(const uint4*)(src);
      p1.q = *(const uint4*)(src + 8);
      #pragma unroll
      for (int i = 0; i < 8; ++i) {
        Blds[(bseg * 16 + i) * 40 + bk]     = p0.h[i];
        Blds[(bseg * 16 + 8 + i) * 40 + bk] = p1.h[i];
      }
      if (kt + 1 < KT) __builtin_prefetch(src + (size_t)32 * N, 0, 1);
    }
#if ASYNC_A
    async_wait0();                    // close this wave's ASYNCcnt before barrier
#endif
    __syncthreads();

    Frag16 af[4], bw[4];
    #pragma unroll
    for (int i = 0; i < 4; ++i) {           // A frag: row=l16, k halves per ISA layout
      const bf_t* base = &Alds[(waveM * 64 + i * 16 + l16) * 40 + lh * 8];
      af[i].q[0] = *(const uint4*)(base);
      af[i].q[1] = *(const uint4*)(base + 16);
    }
    #pragma unroll
    for (int j = 0; j < 4; ++j) {           // B frag: col=l16, k halves per ISA layout
      const bf_t* base = &Blds[(waveN * 64 + j * 16 + l16) * 40 + lh * 8];
      bw[j].q[0] = *(const uint4*)(base);
      bw[j].q[1] = *(const uint4*)(base + 16);
    }
    #pragma unroll
    for (int i = 0; i < 4; ++i)
      #pragma unroll
      for (int j = 0; j < 4; ++j)
        acc[i][j] = wmma_bf16(af[i], bw[j], acc[i][j]);
    __syncthreads();
  }

  #pragma unroll
  for (int i = 0; i < 4; ++i) {
    #pragma unroll
    for (int j = 0; j < 4; ++j) {
      const int col = bn + waveN * 64 + j * 16 + l16;
      const float bv = bias[col];
      #pragma unroll
      for (int r = 0; r < 8; ++r) {
        const int row = bm + waveM * 64 + i * 16 + lh * 8 + r;
        const size_t idx = (size_t)row * N + col;
        float v = acc[i][j][r] + bv;
        if (epi == 1) {
          v += residual[idx];
          outF[idx] = v;
        } else {
          if (epi == 2) v = 0.5f * v * (1.0f + erff(v * 0.70710678118654752f));
          outB[idx] = (bf_t)v;
        }
      }
    }
  }
}

// ---------------------------------------------------------------- fused attention
// grid = (B*NH, T/128); each wave owns 16 query rows; flash-style online softmax.
__global__ __launch_bounds__(256) void attn_kernel(
    const bf_t* __restrict__ Q, const bf_t* __restrict__ Kg,
    const bf_t* __restrict__ V, const float* __restrict__ mask,
    bf_t* __restrict__ ctx)
{
  __shared__ __align__(16) bf_t Vt[64 * 40];       // V transposed: [dim][key] (pad 40)
  __shared__ __align__(16) bf_t Pl[8][16 * 40];    // per-wave P staging

  const int tid  = threadIdx.x;
  const int wid  = tid >> 5;
  const int lane = tid & 31;
  const int l16  = lane & 15;
  const int lh   = lane >> 4;
  const int b = blockIdx.x / NH;
  const int h = blockIdx.x % NH;
  const int q0 = blockIdx.y * 128 + wid * 16;
  const size_t head = (size_t)b * TT * H + (size_t)h * HD;

  Frag16 qf[2];   // Q tile 16x64 as two 16x32 A fragments
  {
    const bf_t* base = Q + head + (size_t)(q0 + l16) * H + lh * 8;
    qf[0].q[0] = *(const uint4*)(base);
    qf[0].q[1] = *(const uint4*)(base + 16);
    qf[1].q[0] = *(const uint4*)(base + 32);
    qf[1].q[1] = *(const uint4*)(base + 48);
  }

  float mi[8], li[8];
  v8f acc[4];
  #pragma unroll
  for (int r = 0; r < 8; ++r) { mi[r] = -1e30f; li[r] = 0.0f; }
  #pragma unroll
  for (int t = 0; t < 4; ++t)
    #pragma unroll
    for (int r = 0; r < 8; ++r) acc[t][r] = 0.0f;

  const int vkey = tid >> 3;
  const int vseg = tid & 7;

  for (int jt = 0; jt < TT / 32; ++jt) {
    { // cooperative V tile transpose into LDS
      Pack8 p;
      p.q = *(const uint4*)(V + head + (size_t)(jt * 32 + vkey) * H + vseg * 8);
      #pragma unroll
      for (int i = 0; i < 8; ++i) Vt[(vseg * 8 + i) * 40 + vkey] = p.h[i];
    }
    __syncthreads();

    v8f S[2];
    #pragma unroll
    for (int nt = 0; nt < 2; ++nt) {       // S = Q * K^T for 16 keys, HD=64 -> 2 WMMAs
      const int key = jt * 32 + nt * 16 + l16;
      const bf_t* kb = Kg + head + (size_t)key * H + lh * 8;
      Frag16 kf;
      v8f s = {0, 0, 0, 0, 0, 0, 0, 0};
      kf.q[0] = *(const uint4*)(kb);
      kf.q[1] = *(const uint4*)(kb + 16);
      s = wmma_bf16(qf[0], kf, s);
      kf.q[0] = *(const uint4*)(kb + 32);
      kf.q[1] = *(const uint4*)(kb + 48);
      s = wmma_bf16(qf[1], kf, s);
      const float am = (1.0f - mask[b * TT + key]) * -10000.0f;
      #pragma unroll
      for (int r = 0; r < 8; ++r) s[r] = s[r] * ATT_SCALE + am;
      S[nt] = s;
    }

    // online softmax: row reductions across the 16-lane halves (wave32 C layout)
    #pragma unroll
    for (int r = 0; r < 8; ++r) {
      float v = fmaxf(S[0][r], S[1][r]);
      v = fmaxf(v, __shfl_xor(v, 1, 32));
      v = fmaxf(v, __shfl_xor(v, 2, 32));
      v = fmaxf(v, __shfl_xor(v, 4, 32));
      v = fmaxf(v, __shfl_xor(v, 8, 32));
      const float mn = fmaxf(mi[r], v);
      const float alpha = __expf(mi[r] - mn);
      mi[r] = mn;
      float p0 = __expf(S[0][r] - mn);
      float p1 = __expf(S[1][r] - mn);
      S[0][r] = p0; S[1][r] = p1;
      float sum = p0 + p1;
      sum += __shfl_xor(sum, 1, 32);
      sum += __shfl_xor(sum, 2, 32);
      sum += __shfl_xor(sum, 4, 32);
      sum += __shfl_xor(sum, 8, 32);
      li[r] = li[r] * alpha + sum;
      #pragma unroll
      for (int t = 0; t < 4; ++t) acc[t][r] *= alpha;
    }

    // reshape P (C layout) -> A fragment via per-wave LDS patch
    bf_t* pw = &Pl[wid][0];
    #pragma unroll
    for (int r = 0; r < 8; ++r) {
      const int prow = lh * 8 + r;
      pw[prow * 40 + l16]      = (bf_t)S[0][r];
      pw[prow * 40 + 16 + l16] = (bf_t)S[1][r];
    }
    Frag16 pf;
    {
      const bf_t* pb = pw + l16 * 40 + lh * 8;
      pf.q[0] = *(const uint4*)(pb);
      pf.q[1] = *(const uint4*)(pb + 16);
    }
    #pragma unroll
    for (int t = 0; t < 4; ++t) {          // ctx += P(16x32) * V(32x64)
      const bf_t* vb = &Vt[(t * 16 + l16) * 40 + lh * 8];
      Frag16 vf;
      vf.q[0] = *(const uint4*)(vb);
      vf.q[1] = *(const uint4*)(vb + 16);
      acc[t] = wmma_bf16(pf, vf, acc[t]);
    }
    __syncthreads();
  }

  #pragma unroll
  for (int t = 0; t < 4; ++t) {
    const int d = t * 16 + l16;
    #pragma unroll
    for (int r = 0; r < 8; ++r) {
      const int row = q0 + lh * 8 + r;
      ctx[head + (size_t)row * H + d] = (bf_t)(acc[t][r] / li[r]);
    }
  }
}

// ---------------------------------------------------------------- layernorm (row=768)
__global__ __launch_bounds__(256) void ln_kernel(
    const float* __restrict__ x, const float* __restrict__ g,
    const float* __restrict__ bt, float* __restrict__ outF,
    bf_t* __restrict__ outB)
{
  const int row = blockIdx.x;
  const int tid = threadIdx.x;
  const float* xr = x + (size_t)row * H;
  float e0 = xr[tid], e1 = xr[tid + 256], e2 = xr[tid + 512];
  __shared__ float rs[256], rq[256];
  rs[tid] = e0 + e1 + e2;
  rq[tid] = e0 * e0 + e1 * e1 + e2 * e2;
  __syncthreads();
  for (int o = 128; o > 0; o >>= 1) {
    if (tid < o) { rs[tid] += rs[tid + o]; rq[tid] += rq[tid + o]; }
    __syncthreads();
  }
  const float mean = rs[0] * (1.0f / H);
  const float var  = rq[0] * (1.0f / H) - mean * mean;
  const float inv  = rsqrtf(var + 1e-12f);
  float e[3] = {e0, e1, e2};
  #pragma unroll
  for (int i = 0; i < 3; ++i) {
    const int c = tid + i * 256;
    const float y = (e[i] - mean) * inv * g[c] + bt[c];
    outF[(size_t)row * H + c] = y;
    outB[(size_t)row * H + c] = (bf_t)y;
  }
}

// ---------------------------------------------------------------- host
static inline void launch_cvt(const float* s, bf_t* d, size_t n, hipStream_t st) {
  cvt_kernel<<<(unsigned)((n + 255) / 256), 256, 0, st>>>(s, d, n);
}

extern "C" void kernel_launch(void* const* d_in, const int* in_sizes, int n_in,
                              void* d_out, int out_size, void* d_ws, size_t ws_size,
                              hipStream_t stream)
{
  const float* hs  = (const float*)d_in[0];
  const float* am  = (const float*)d_in[1];
  const float* Wq  = (const float*)d_in[2];
  const float* bq  = (const float*)d_in[3];
  const float* Wk  = (const float*)d_in[4];
  const float* bk  = (const float*)d_in[5];
  const float* Wv  = (const float*)d_in[6];
  const float* bv  = (const float*)d_in[7];
  const float* Wao = (const float*)d_in[8];
  const float* bao = (const float*)d_in[9];
  const float* g1  = (const float*)d_in[10];
  const float* b1  = (const float*)d_in[11];
  const float* Wi  = (const float*)d_in[12];
  const float* bi  = (const float*)d_in[13];
  const float* Wo  = (const float*)d_in[14];
  const float* bo  = (const float*)d_in[15];
  const float* g2  = (const float*)d_in[16];
  const float* b2  = (const float*)d_in[17];

  char* ws = (char*)d_ws;
  size_t off = 0;
  auto alloc = [&](size_t bytes) -> void* {
    void* p = ws + off;
    off += (bytes + 255) & ~(size_t)255;
    return p;
  };

  const size_t WHH = (size_t)LYR * H * H;
  const size_t WHF = (size_t)LYR * H * FFD;

  bf_t* wq_b  = (bf_t*)alloc(WHH * 2);
  bf_t* wk_b  = (bf_t*)alloc(WHH * 2);
  bf_t* wv_b  = (bf_t*)alloc(WHH * 2);
  bf_t* wao_b = (bf_t*)alloc(WHH * 2);
  bf_t* wi_b  = (bf_t*)alloc(WHF * 2);
  bf_t* wo_b  = (bf_t*)alloc(WHF * 2);
  bf_t* x_b   = (bf_t*)alloc((size_t)BT * H * 2);
  bf_t* q_b   = (bf_t*)alloc((size_t)BT * H * 2);
  bf_t* k_b   = (bf_t*)alloc((size_t)BT * H * 2);
  bf_t* v_b   = (bf_t*)alloc((size_t)BT * H * 2);
  bf_t* ctx_b = (bf_t*)alloc((size_t)BT * H * 2);
  bf_t* a1_b  = (bf_t*)alloc((size_t)BT * H * 2);
  bf_t* ff_b  = (bf_t*)alloc((size_t)BT * FFD * 2);
  float* x_f   = (float*)alloc((size_t)BT * H * 4);
  float* pre1  = (float*)alloc((size_t)BT * H * 4);
  float* a1_f  = (float*)alloc((size_t)BT * H * 4);
  float* pre2  = (float*)alloc((size_t)BT * H * 4);

  // one-time per-launch conversions (deterministic, graph-capture safe)
  launch_cvt(Wq,  wq_b,  WHH, stream);
  launch_cvt(Wk,  wk_b,  WHH, stream);
  launch_cvt(Wv,  wv_b,  WHH, stream);
  launch_cvt(Wao, wao_b, WHH, stream);
  launch_cvt(Wi,  wi_b,  WHF, stream);
  launch_cvt(Wo,  wo_b,  WHF, stream);
  launch_cvt(hs,  x_b, (size_t)BT * H, stream);
  (void)hipMemcpyAsync(x_f, hs, (size_t)BT * H * 4, hipMemcpyDeviceToDevice, stream);

  const dim3 blk(256, 1, 1);
  const dim3 gH(H / 128, BT / 256, 1);    // N=768 GEMMs:  (6, 16)
  const dim3 gF(FFD / 128, BT / 256, 1);  // N=3072 GEMM:  (24, 16)
  const dim3 gA(BB * NH, TT / 128, 1);    // attention:    (96, 4)

  for (int l = 0; l < LYR; ++l) {
    const bf_t* wq = wq_b  + (size_t)l * H * H;
    const bf_t* wk = wk_b  + (size_t)l * H * H;
    const bf_t* wv = wv_b  + (size_t)l * H * H;
    const bf_t* wa = wao_b + (size_t)l * H * H;
    const bf_t* wi = wi_b  + (size_t)l * H * FFD;
    const bf_t* wo = wo_b  + (size_t)l * FFD * H;

    gemm_bf16_kernel<<<gH, blk, 0, stream>>>(x_b, wq, bq + l * H, nullptr,
                                             nullptr, q_b, BT, H, H, 0);
    gemm_bf16_kernel<<<gH, blk, 0, stream>>>(x_b, wk, bk + l * H, nullptr,
                                             nullptr, k_b, BT, H, H, 0);
    gemm_bf16_kernel<<<gH, blk, 0, stream>>>(x_b, wv, bv + l * H, nullptr,
                                             nullptr, v_b, BT, H, H, 0);
    attn_kernel<<<gA, blk, 0, stream>>>(q_b, k_b, v_b, am, ctx_b);
    gemm_bf16_kernel<<<gH, blk, 0, stream>>>(ctx_b, wa, bao + l * H, x_f,
                                             pre1, nullptr, BT, H, H, 1);
    ln_kernel<<<BT, blk, 0, stream>>>(pre1, g1 + l * H, b1 + l * H, a1_f, a1_b);
    gemm_bf16_kernel<<<gF, blk, 0, stream>>>(a1_b, wi, bi + l * FFD, nullptr,
                                             nullptr, ff_b, BT, FFD, H, 2);
    gemm_bf16_kernel<<<gH, blk, 0, stream>>>(ff_b, wo, bo + l * H, a1_f,
                                             pre2, nullptr, BT, H, FFD, 1);
    float* lnOut = (l == LYR - 1) ? (float*)d_out : x_f;
    ln_kernel<<<BT, blk, 0, stream>>>(pre2, g2 + l * H, b2 + l * H, lnOut, x_b);
  }
}
